// AdaptedAttention_4896262718044
// MI455X (gfx1250) — matrix operand order repaired and checked by reading, not compiled, not run on hardware
//
#include <hip/hip_runtime.h>
#include <hip/hip_bf16.h>
#include <stdint.h>
#include <math.h>

// Problem constants (match reference)
#define B_   4
#define S_   2048
#define HID_ 4096
#define H_   32
#define D_   128
#define L_   64

typedef __attribute__((ext_vector_type(16))) __bf16 v16bf;
typedef __attribute__((ext_vector_type(8)))  float  v8f;

static __device__ __forceinline__ v8f zero8() {
  v8f z = {0.f, 0.f, 0.f, 0.f, 0.f, 0.f, 0.f, 0.f};
  return z;
}

static __device__ __forceinline__ v8f wmma_bf16(v16bf a, v16bf b, v8f c) {
  // (neg_a, A, neg_b, B, c_mod, C, reuse_a, reuse_b)
  return __builtin_amdgcn_wmma_f32_16x16x32_bf16(false, a, false, b, (short)0, c,
                                                 false, false);
}

// CDNA5 async global->LDS copy (ASYNCcnt-tracked), 16B per lane.
static __device__ __forceinline__ void async_load_b128(uint32_t lds_byte_addr,
                                                       const void* gptr) {
  asm volatile("global_load_async_to_lds_b128 %0, %1, off"
               :: "v"(lds_byte_addr), "v"((uint64_t)(uintptr_t)gptr)
               : "memory");
}
static __device__ __forceinline__ void wait_async0() {
  asm volatile("s_wait_asynccnt 0" ::: "memory");
}

// ---- WMMA fragment packing (wave32, 16x16x32 bf16) -------------------------
// A (16x32, M-major in LDS, row stride ld):
//   lanes 0-15: M=lane, K 0..15; lanes 16-31: M=lane-16, K 16..31.
static __device__ __forceinline__ v16bf frag_a(const __bf16* base, int ld, int lane) {
  const int m  = lane & 15;
  const int kh = (lane & 16) ? 16 : 0;
  const __bf16* p = base + m * ld + kh;
  v16bf a;
#pragma unroll
  for (int e = 0; e < 16; ++e) {
    a[e] = p[e];
  }
  return a;
}

// B (32x16) from N-major storage (Bs[n][k], row stride ld)
static __device__ __forceinline__ v16bf frag_b_n(const __bf16* base, int ld, int lane) {
  const int n  = lane & 15;
  const int kh = (lane & 16) ? 16 : 0;
  const __bf16* p = base + n * ld + kh;
  v16bf b;
#pragma unroll
  for (int e = 0; e < 16; ++e) b[e] = p[e];
  return b;
}

// B (32x16) from K-major storage (Bs[k][n], row stride ld)
static __device__ __forceinline__ v16bf frag_b_k(const __bf16* base, int ld, int lane) {
  const int n  = lane & 15;
  const int kh = (lane & 16) ? 16 : 0;
  const __bf16* p = base + kh * ld + n;
  v16bf b;
#pragma unroll
  for (int e = 0; e < 16; ++e) b[e] = p[e * ld];
  return b;
}

// ---------------------------------------------------------------------------
// Kernel 0: fp32 -> bf16 conversion (grid-stride)
__global__ __launch_bounds__(256) void cvt_f32_bf16(const float* __restrict__ src,
                                                    __bf16* __restrict__ dst, int n) {
  int i = blockIdx.x * blockDim.x + threadIdx.x;
  const int stride = gridDim.x * blockDim.x;
  for (; i < n; i += stride) dst[i] = (__bf16)src[i];
}

// ---------------------------------------------------------------------------
// Kernel 1: adapter K/V projections.  grid = (H, 2); block computes one head's
// [L=64, D=128] = prompt[64,4096] @ W_head^T via bf16 WMMA, f32 accumulate.
// (fp32 source needs conversion, so stage through VGPRs here.)
__global__ __launch_bounds__(256) void adapter_kv_kernel(
    const float* __restrict__ prompt, const float* __restrict__ Wk,
    const float* __restrict__ Wv, __bf16* __restrict__ Kb, __bf16* __restrict__ Vb) {
  const int h  = blockIdx.x;
  const int kv = blockIdx.y;
  const float* __restrict__ W = kv ? Wv : Wk;
  __bf16* __restrict__ out = (kv ? Vb : Kb) + h * L_ * D_;

  __shared__ __bf16 As[L_][72];   // prompt chunk [64][64] bf16
  __shared__ __bf16 Bs[D_][72];   // W_head chunk [128][64] bf16 (N-major)

  const int tid = threadIdx.x, w = tid >> 5, lane = tid & 31;
  const int mt = w & 3, ntg = w >> 2;   // C = 4x8 tiles of 16x16; 4 tiles/wave

  v8f acc[4];
#pragma unroll
  for (int j = 0; j < 4; ++j) acc[j] = zero8();

  for (int kc = 0; kc < HID_; kc += 64) {
    for (int idx = tid; idx < 64 * 64; idx += 256) {
      const int r = idx >> 6, c = idx & 63;
      As[r][c] = (__bf16)prompt[r * HID_ + kc + c];
    }
    for (int idx = tid; idx < 128 * 64; idx += 256) {
      const int r = idx >> 6, c = idx & 63;
      Bs[r][c] = (__bf16)W[(h * D_ + r) * HID_ + kc + c];
    }
    __syncthreads();
#pragma unroll
    for (int ks = 0; ks < 2; ++ks) {
      const int k0 = ks * 32;
      const v16bf a = frag_a(&As[mt * 16][k0], 72, lane);
#pragma unroll
      for (int j = 0; j < 4; ++j) {
        const v16bf b = frag_b_n(&Bs[(ntg * 4 + j) * 16][k0], 72, lane);
        acc[j] = wmma_bf16(a, b, acc[j]);
      }
    }
    __syncthreads();
  }
#pragma unroll
  for (int j = 0; j < 4; ++j) {
    const int n  = (ntg * 4 + j) * 16 + (lane & 15);
    const int mb = mt * 16 + ((lane & 16) ? 8 : 0);
#pragma unroll
    for (int r = 0; r < 8; ++r) out[(mb + r) * D_ + n] = (__bf16)acc[j][r];
  }
}

// ---------------------------------------------------------------------------
// Kernel 2: fused Q-proj + RoPE + adapter attention for one (64-row S tile, head).
// grid = (B*S/64, H).  Q-proj K-loop uses double-buffered async global->LDS.
__global__ __launch_bounds__(256) void fused_attn_kernel(
    const __bf16* __restrict__ Xb, const __bf16* __restrict__ Wqb,
    const __bf16* __restrict__ Kb, const __bf16* __restrict__ Vb,
    const float* __restrict__ gatep, __bf16* __restrict__ AdOut) {
  const int b    = blockIdx.x >> 5;          // S_/64 == 32 tiles per batch
  const int st   = blockIdx.x & 31;
  const int s0   = st * 64;
  const int row0 = b * S_ + s0;              // row into [B*S, HID]
  const int h    = blockIdx.y;

  // LDS overlay (55296 B):
  //  phase 1: two stage buffers @ {0, 27648}: As [64][72]bf16 @ +0,
  //           Bs [128][72]bf16 @ +9216 (row stride 144 B = 9 x 16B chunks)
  //  later phases alias the dead buffers (see offsets below).
  __shared__ __align__(16) char smem[55296];
  float*  Qs  = (float*) (smem);            // [64][132] fp32 Q tile
  __bf16* Qbf = (__bf16*)(smem + 33792);    // [64][136] RoPE'd Q (bf16)
  __bf16* Ks  = (__bf16*)(smem);            // [64][136] head K
  float*  Ss  = (float*) (smem + 33792);    // [64][68]  scores (aliases dead Qbf)
  __bf16* Vs  = (__bf16*)(smem);            // [64][136] head V (aliases dead Ks)
  __bf16* Pb  = (__bf16*)(smem + 17408);    // [64][72]  gate*softmax (bf16)

  const int tid = threadIdx.x, w = tid >> 5, lane = tid & 31;
  const int mt = w & 3, ntg = w >> 2;
  const float gate = gatep[0];
  const uint32_t smem0 = (uint32_t)(uintptr_t)(void*)smem;  // LDS byte base

  // ---- Phase 1: Q = X_tile @ Wq_head^T (64x128, K=4096), async pipelined ----
  auto issue_slab = [&](int buf, int kc) {
    const uint32_t asb = smem0 + buf * 27648;
    const uint32_t bsb = asb + 9216;
#pragma unroll
    for (int q = 0; q < 2; ++q) {            // X: 512 16B chunks (64 x 64 bf16)
      const int c = tid + 256 * q;
      const int r = c >> 3, col = c & 7;
      async_load_b128(asb + r * 144 + col * 16,
                      Xb + (row0 + r) * HID_ + kc + col * 8);
    }
#pragma unroll
    for (int q = 0; q < 4; ++q) {            // Wq head: 1024 16B chunks
      const int c = tid + 256 * q;
      const int r = c >> 3, col = c & 7;
      async_load_b128(bsb + r * 144 + col * 16,
                      Wqb + (h * D_ + r) * HID_ + kc + col * 8);
    }
  };

  v8f qacc[4];
#pragma unroll
  for (int j = 0; j < 4; ++j) qacc[j] = zero8();

  issue_slab(0, 0);
  wait_async0();
  __syncthreads();
  for (int it = 0; it < 64; ++it) {
    const int cur = it & 1;
    if (it + 1 < 64) issue_slab(cur ^ 1, (it + 1) * 64);
    const __bf16* As = (const __bf16*)(smem + cur * 27648);
    const __bf16* Bs = As + 4608;  // +9216 bytes
#pragma unroll
    for (int ks = 0; ks < 2; ++ks) {
      const int k0 = ks * 32;
      const v16bf a = frag_a(As + (mt * 16) * 72 + k0, 72, lane);
#pragma unroll
      for (int j = 0; j < 4; ++j) {
        const v16bf bf = frag_b_n(Bs + ((ntg * 4 + j) * 16) * 72 + k0, 72, lane);
        qacc[j] = wmma_bf16(a, bf, qacc[j]);
      }
    }
    if (it + 1 < 64) wait_async0();
    __syncthreads();
  }

  // ---- Phase 2: spill fp32 Q tile to LDS (C-matrix layout) ----
#pragma unroll
  for (int j = 0; j < 4; ++j) {
    const int n  = (ntg * 4 + j) * 16 + (lane & 15);
    const int mb = mt * 16 + ((lane & 16) ? 8 : 0);
#pragma unroll
    for (int r = 0; r < 8; ++r) Qs[(mb + r) * 132 + n] = qacc[j][r];
  }
  __syncthreads();

  // ---- Phase 3: RoPE (fold 1/sqrt(D)) -> bf16 Q ----
  for (int idx = tid; idx < 64 * 128; idx += 256) {
    const int m = idx >> 7, d = idx & 127, dd = d & 63;
    const float ang = (float)(s0 + m) * exp2f(-(float)dd * (13.2877123795494f / 64.f));
    float sn, cs;
    sincosf(ang, &sn, &cs);
    const float q  = Qs[m * 132 + d];
    const float qo = (d < 64) ? (q * cs - Qs[m * 132 + d + 64] * sn)
                              : (q * cs + Qs[m * 132 + d - 64] * sn);
    Qbf[m * 136 + d] = (__bf16)(qo * 0.0883883476483184f);  // 1/sqrt(128)
  }
  __syncthreads();

  // load head K (bf16) into LDS (aliases dead Qs region)
  {
    const __bf16* Kh = Kb + h * L_ * D_;
    for (int idx = tid; idx < 64 * 64; idx += 256) {
      const int r = idx >> 6, cw = idx & 63;
      ((uint32_t*)(Ks + r * 136))[cw] = ((const uint32_t*)(Kh + r * D_))[cw];
    }
  }
  __syncthreads();

  // ---- Phase 4: scores = Qr @ K^T  (64x64, K=128) ----
  v8f sacc[2];
  sacc[0] = zero8(); sacc[1] = zero8();
#pragma unroll
  for (int ks = 0; ks < 4; ++ks) {
    const int k0 = ks * 32;
    const v16bf a = frag_a(Qbf + (mt * 16) * 136 + k0, 136, lane);
#pragma unroll
    for (int j = 0; j < 2; ++j) {
      const int nt = (w >> 2) * 2 + j;
      const v16bf bfr = frag_b_n(Ks + (nt * 16) * 136 + k0, 136, lane);
      sacc[j] = wmma_bf16(a, bfr, sacc[j]);
    }
  }
  __syncthreads();   // Qbf & Ks now dead; safe to alias with Ss / Vs

  // scores -> Ss (aliases Qbf), head V -> Vs (aliases Ks)
#pragma unroll
  for (int j = 0; j < 2; ++j) {
    const int nt = (w >> 2) * 2 + j;
    const int n  = nt * 16 + (lane & 15);
    const int mb = mt * 16 + ((lane & 16) ? 8 : 0);
#pragma unroll
    for (int r = 0; r < 8; ++r) Ss[(mb + r) * 68 + n] = sacc[j][r];
  }
  {
    const __bf16* Vh = Vb + h * L_ * D_;
    for (int idx = tid; idx < 64 * 64; idx += 256) {
      const int r = idx >> 6, cw = idx & 63;
      ((uint32_t*)(Vs + r * 136))[cw] = ((const uint32_t*)(Vh + r * D_))[cw];
    }
  }
  __syncthreads();

  // ---- Phase 5: row softmax * gate -> bf16 P ----
  if (tid < 64) {
    const int m = tid;
    float mx = -1e30f;
    for (int l = 0; l < L_; ++l) mx = fmaxf(mx, Ss[m * 68 + l]);
    float sum = 0.f;
    for (int l = 0; l < L_; ++l) {
      const float e = expf(Ss[m * 68 + l] - mx);
      Ss[m * 68 + l] = e;
      sum += e;
    }
    const float inv = gate / sum;
    for (int l = 0; l < L_; ++l) Pb[m * 72 + l] = (__bf16)(Ss[m * 68 + l] * inv);
  }
  __syncthreads();

  // ---- Phase 6: adapter_out = P @ V  (64x128, K=64) ----
  v8f oacc[4];
#pragma unroll
  for (int j = 0; j < 4; ++j) oacc[j] = zero8();
#pragma unroll
  for (int ks = 0; ks < 2; ++ks) {
    const int k0 = ks * 32;
    const v16bf a = frag_a(Pb + (mt * 16) * 72 + k0, 72, lane);
#pragma unroll
    for (int j = 0; j < 4; ++j) {
      const int nt = ntg * 4 + j;
      const v16bf bf = frag_b_k(Vs + k0 * 136 + nt * 16, 136, lane);
      oacc[j] = wmma_bf16(a, bf, oacc[j]);
    }
  }
  // ---- Phase 7: store bf16 adapter head output ----
#pragma unroll
  for (int j = 0; j < 4; ++j) {
    const int n  = (ntg * 4 + j) * 16 + (lane & 15);
    const int mb = mt * 16 + ((lane & 16) ? 8 : 0);
#pragma unroll
    for (int r = 0; r < 8; ++r)
      AdOut[(row0 + mb + r) * HID_ + h * D_ + n] = (__bf16)oacc[j][r];
  }
}

// ---------------------------------------------------------------------------
// Kernel 3: out = base + AdOut @ Wo^T.  grid = (B*S/64, HID/64), 64x64 C tiles.
// Async double-buffered K-loop.
__global__ __launch_bounds__(256) void oproj_kernel(
    const __bf16* __restrict__ Ab, const __bf16* __restrict__ Wob,
    const float* __restrict__ base, float* __restrict__ out) {
  const int row0 = blockIdx.x * 64;
  const int n0   = blockIdx.y * 64;

  // two stage buffers @ {0, 18432}: As [64][72]bf16 @ +0, Bs [64][72]bf16 @ +9216
  __shared__ __align__(16) char smem[36864];
  const int tid = threadIdx.x, w = tid >> 5, lane = tid & 31;
  const int mt = w & 3;
  const uint32_t smem0 = (uint32_t)(uintptr_t)(void*)smem;

  auto issue_slab = [&](int buf, int kc) {
    const uint32_t asb = smem0 + buf * 18432;
    const uint32_t bsb = asb + 9216;
#pragma unroll
    for (int q = 0; q < 2; ++q) {            // A: 512 16B chunks
      const int c = tid + 256 * q;
      const int r = c >> 3, col = c & 7;
      async_load_b128(asb + r * 144 + col * 16,
                      Ab + (row0 + r) * HID_ + kc + col * 8);
    }
#pragma unroll
    for (int q = 0; q < 2; ++q) {            // Wo rows n0..n0+63
      const int c = tid + 256 * q;
      const int r = c >> 3, col = c & 7;
      async_load_b128(bsb + r * 144 + col * 16,
                      Wob + (n0 + r) * HID_ + kc + col * 8);
    }
  };

  v8f acc[2];
  acc[0] = zero8(); acc[1] = zero8();

  issue_slab(0, 0);
  wait_async0();
  __syncthreads();
  for (int it = 0; it < 64; ++it) {
    const int cur = it & 1;
    if (it + 1 < 64) issue_slab(cur ^ 1, (it + 1) * 64);
    const __bf16* As = (const __bf16*)(smem + cur * 18432);
    const __bf16* Bs = As + 4608;  // +9216 bytes
#pragma unroll
    for (int ks = 0; ks < 2; ++ks) {
      const int k0 = ks * 32;
      const v16bf a = frag_a(As + (mt * 16) * 72 + k0, 72, lane);
#pragma unroll
      for (int j = 0; j < 2; ++j) {
        const int nt = (w >> 2) * 2 + j;
        const v16bf bf = frag_b_n(Bs + (nt * 16) * 72 + k0, 72, lane);
        acc[j] = wmma_bf16(a, bf, acc[j]);
      }
    }
    if (it + 1 < 64) wait_async0();
    __syncthreads();
  }
#pragma unroll
  for (int j = 0; j < 2; ++j) {
    const int nt = (w >> 2) * 2 + j;
    const int n  = n0 + nt * 16 + (lane & 15);
    const int mb = row0 + mt * 16 + ((lane & 16) ? 8 : 0);
#pragma unroll
    for (int r = 0; r < 8; ++r)
      out[(mb + r) * HID_ + n] = base[(mb + r) * HID_ + n] + acc[j][r];
  }
}

// ---------------------------------------------------------------------------
extern "C" void kernel_launch(void* const* d_in, const int* in_sizes, int n_in,
                              void* d_out, int out_size, void* d_ws, size_t ws_size,
                              hipStream_t stream) {
  const float* hs     = (const float*)d_in[0];
  const float* base   = (const float*)d_in[1];
  const float* Wq     = (const float*)d_in[2];
  const float* Wk     = (const float*)d_in[3];
  const float* Wv     = (const float*)d_in[4];
  const float* Wo     = (const float*)d_in[5];
  const float* prompt = (const float*)d_in[6];
  const float* gate   = (const float*)d_in[7];
  // d_in[8] = position_ids == arange(S) broadcast; position == s index, used directly.
  float* out = (float*)d_out;

  // Workspace layout (~193 MB total, assumed <= ws_size):
  char* ws = (char*)d_ws;
  __bf16* Xb  = (__bf16*)(ws);                 //  64 MB  bf16 hidden_states
  __bf16* Wqb = (__bf16*)(ws + 67108864);      //  32 MB  bf16 Wq
  __bf16* Wob = (__bf16*)(ws + 100663296);     //  32 MB  bf16 Wo
  __bf16* Kb  = (__bf16*)(ws + 134217728);     // 512 KB  bf16 adapter K [H,L,D]
  __bf16* Vb  = (__bf16*)(ws + 134742016);     // 512 KB  bf16 adapter V [H,L,D]
  __bf16* Ab  = (__bf16*)(ws + 135266304);     //  64 MB  bf16 adapter output

  cvt_f32_bf16<<<2048, 256, 0, stream>>>(hs, Xb, B_ * S_ * HID_);
  cvt_f32_bf16<<<1024, 256, 0, stream>>>(Wq, Wqb, HID_ * HID_);
  cvt_f32_bf16<<<1024, 256, 0, stream>>>(Wo, Wob, HID_ * HID_);
  adapter_kv_kernel<<<dim3(32, 2), 256, 0, stream>>>(prompt, Wk, Wv, Kb, Vb);
  fused_attn_kernel<<<dim3(128, 32), 256, 0, stream>>>(Xb, Wqb, Kb, Vb, gate, Ab);
  oproj_kernel<<<dim3(128, 64), 256, 0, stream>>>(Ab, Wob, base, out);
}
